// _Expert_15564961480892
// MI455X (gfx1250) — matrix operand order repaired and checked by reading, not compiled, not run on hardware
//
#include <hip/hip_runtime.h>
#include <math.h>

// ---------------------------------------------------------------------------
// Grouped expert FFN for MI455X (gfx1250, wave32).
//   pass1: h   = gelu(x @ w1[e]^T + b1[e])     M=1024/expert, N=4096, K=1024
//   pass2: out =      h @ w2[e]^T + b2[e]      M=1024/expert, N=1024, K=4096
// Split-precision bf16x3: x = hi + lo (bf16 each), A*B ~= Ahi*Bhi + Ahi*Blo
// + Alo*Bhi with f32 WMMA accumulation -> ~2^-16 relative error at bf16
// matrix-op rates. Fast path pre-splits all operands into bf16 hi/lo planes
// so the GEMM inner loop is pure loads + v_wmma (no per-iteration cvt VALU).
// ---------------------------------------------------------------------------

typedef __attribute__((ext_vector_type(16))) __bf16 v16bf;
typedef __attribute__((ext_vector_type(8)))  __bf16 v8bf;
typedef __attribute__((ext_vector_type(4)))  __bf16 v4bf;
typedef __attribute__((ext_vector_type(8)))  float  v8f;
typedef __attribute__((ext_vector_type(4)))  float  f4;

#define TILES_M 2                    // 16-row tiles per wave
#define TILES_N 4                    // 16-col tiles per wave
#define WAVES_M 4
#define WAVES_N 2
#define BLOCK_M (WAVES_M * TILES_M * 16)   // 128 rows per workgroup
#define BLOCK_N (WAVES_N * TILES_N * 16)   // 128 cols per workgroup

// ============================ pre-split kernel =============================
// src f32 -> hi/lo bf16 planes, vectorized x4, grid-stride.
__global__ __launch_bounds__(256)
void split_f32_to_bf16(const float* __restrict__ src,
                       __bf16* __restrict__ hi, __bf16* __restrict__ lo,
                       long long n4) {
  long long i = (long long)blockIdx.x * blockDim.x + threadIdx.x;
  const long long stride = (long long)gridDim.x * blockDim.x;
  for (; i < n4; i += stride) {
    f4 x = ((const f4*)src)[i];
    v4bf h, l;
#pragma unroll
    for (int j = 0; j < 4; ++j) {
      __bf16 hv = (__bf16)x[j];
      h[j] = hv;
      l[j] = (__bf16)(x[j] - (float)hv);
    }
    ((v4bf*)hi)[i] = h;
    ((v4bf*)lo)[i] = l;
  }
}

// ====================== fragment loads (pre-split bf16) ====================
// A fragment (16x32, MxK): lane row=l16 holds K=kk+hl*16+i in elems i=0..15
// (hl = lane>>4), matching the B fragment order.
__device__ __forceinline__ v16bf load_frag_a_bf(const __bf16* __restrict__ P,
                                                int ldx, int row, int kk, int hl) {
  const __bf16* p = P + (size_t)row * ldx + kk + hl * 16;
  v8bf u0 = *(const v8bf*)(p);
  v8bf u1 = *(const v8bf*)(p + 8);
  v16bf r;
#pragma unroll
  for (int i = 0; i < 8; ++i) { r[i] = u0[i]; r[8 + i] = u1[i]; }
  return r;
}

// B fragment (32x16, KxN), B[k][n] = W[n][k]: lane col=l16, elems i=0..15 hold
// K = kk + hl*16 + i -> 16 contiguous bf16 (32B, one aligned vector load).
__device__ __forceinline__ v16bf load_frag_b_bf(const __bf16* __restrict__ P,
                                                int ldw, int col, int kk, int hl) {
  return *(const v16bf*)(P + (size_t)col * ldw + kk + hl * 16);
}

// ===================== GEMM on pre-split hi/lo planes ======================
__global__ __launch_bounds__(256, 1)
void grouped_gemm_bf16x3_pre(const __bf16* __restrict__ Xhi,  // [Ttot, K]
                             const __bf16* __restrict__ Xlo,
                             const __bf16* __restrict__ Whi,  // [E, Ntot, K]
                             const __bf16* __restrict__ Wlo,
                             const float* __restrict__ bias,  // [E, Ntot]
                             float* __restrict__ OutF,        // pass2: [Ttot,Ntot]
                             __bf16* __restrict__ OutHi,      // pass1: split h
                             __bf16* __restrict__ OutLo,
                             const int* __restrict__ counts,
                             int K, int Ntot, int Ttot, int act_gelu) {
  const int e = blockIdx.z;
  int segBase = 0;
  for (int i = 0; i < e; ++i) segBase += counts[i];
  const int cnt = counts[e];

  const int lane = threadIdx.x & 31;
  const int wave = threadIdx.x >> 5;
  const int hl   = lane >> 4;
  const int l16  = lane & 15;
  const int wm   = wave & (WAVES_M - 1);
  const int wn   = wave >> 2;

  const int rowBase = segBase + blockIdx.y * BLOCK_M + wm * (TILES_M * 16);
  const int colBase = blockIdx.x * BLOCK_N + wn * (TILES_N * 16);
  const size_t wOff = (size_t)e * Ntot * K;
  const __bf16* WeHi = Whi + wOff;
  const __bf16* WeLo = Wlo + wOff;

  v8f acc[TILES_M][TILES_N];
  const v8f vzero = {};
#pragma unroll
  for (int mt = 0; mt < TILES_M; ++mt)
#pragma unroll
    for (int nt = 0; nt < TILES_N; ++nt) acc[mt][nt] = vzero;

  int arow[TILES_M];
#pragma unroll
  for (int mt = 0; mt < TILES_M; ++mt) {
    int r = rowBase + mt * 16 + l16;
    arow[mt] = (r >= Ttot) ? (Ttot - 1) : r;   // clamp; masked at store time
  }

  for (int kk = 0; kk < K; kk += 32) {
    v16bf ahi[TILES_M], alo[TILES_M];
#pragma unroll
    for (int mt = 0; mt < TILES_M; ++mt) {
      ahi[mt] = load_frag_a_bf(Xhi, K, arow[mt], kk, hl);
      alo[mt] = load_frag_a_bf(Xlo, K, arow[mt], kk, hl);
    }
    v16bf bhi[TILES_N], blo[TILES_N];
#pragma unroll
    for (int nt = 0; nt < TILES_N; ++nt) {
      int c = colBase + nt * 16 + l16;
      bhi[nt] = load_frag_b_bf(WeHi, K, c, kk, hl);
      blo[nt] = load_frag_b_bf(WeLo, K, c, kk, hl);
    }
#pragma unroll
    for (int mt = 0; mt < TILES_M; ++mt)
#pragma unroll
      for (int nt = 0; nt < TILES_N; ++nt) {
        acc[mt][nt] = __builtin_amdgcn_wmma_f32_16x16x32_bf16(
            false, ahi[mt], false, bhi[nt], (short)0, acc[mt][nt], false, false);
        acc[mt][nt] = __builtin_amdgcn_wmma_f32_16x16x32_bf16(
            false, ahi[mt], false, blo[nt], (short)0, acc[mt][nt], false, false);
        acc[mt][nt] = __builtin_amdgcn_wmma_f32_16x16x32_bf16(
            false, alo[mt], false, bhi[nt], (short)0, acc[mt][nt], false, false);
      }
  }

  // Epilogue: C/D layout — VGPR g holds row 8*hl + g, col = l16.
  const int rowEnd = segBase + cnt;
#pragma unroll
  for (int mt = 0; mt < TILES_M; ++mt)
#pragma unroll
    for (int nt = 0; nt < TILES_N; ++nt) {
      const int col = colBase + nt * 16 + l16;
      const float bv = bias[(size_t)e * Ntot + col];
#pragma unroll
      for (int g = 0; g < 8; ++g) {
        const int row = rowBase + mt * 16 + hl * 8 + g;
        if (row < rowEnd) {
          float v = acc[mt][nt][g] + bv;
          if (act_gelu)
            v = 0.5f * v * (1.0f + erff(v * 0.70710678118654752f));
          const size_t idx = (size_t)row * Ntot + col;
          if (OutF) {
            OutF[idx] = v;
          } else {
            __bf16 hv = (__bf16)v;          // split h for pass-2 A operand
            OutHi[idx] = hv;
            OutLo[idx] = (__bf16)(v - (float)hv);
          }
        }
      }
    }
}

// ================= fallback: in-register split from f32 ===================
__device__ __forceinline__ void split16(const f4 u[4], v16bf& hi, v16bf& lo) {
#pragma unroll
  for (int i = 0; i < 16; ++i) {
    float x  = u[i >> 2][i & 3];
    __bf16 h = (__bf16)x;
    hi[i] = h;
    lo[i] = (__bf16)(x - (float)h);
  }
}

__device__ __forceinline__ void load_frag_a(const float* __restrict__ X, int ldx,
                                            int row, int kk, int hl,
                                            v16bf& hi, v16bf& lo) {
  const float* p = X + (size_t)row * ldx + kk + hl * 16;
  f4 u[4];
  u[0] = *(const f4*)(p);
  u[1] = *(const f4*)(p + 4);
  u[2] = *(const f4*)(p + 8);
  u[3] = *(const f4*)(p + 12);
  split16(u, hi, lo);
}

__device__ __forceinline__ void load_frag_b(const float* __restrict__ W, int ldw,
                                            int col, int kk, int hl,
                                            v16bf& hi, v16bf& lo) {
  const float* p = W + (size_t)col * ldw + kk + hl * 16;
  f4 u[4];
  u[0] = *(const f4*)(p);
  u[1] = *(const f4*)(p + 4);
  u[2] = *(const f4*)(p + 8);
  u[3] = *(const f4*)(p + 12);
  split16(u, hi, lo);
}

__global__ __launch_bounds__(256, 1)
void grouped_gemm_bf16x3(const float* __restrict__ X,
                         const float* __restrict__ W,
                         const float* __restrict__ bias,
                         float* __restrict__ Out,
                         const int* __restrict__ counts,
                         int K, int Ntot, int Ttot, int act_gelu) {
  const int e = blockIdx.z;
  int segBase = 0;
  for (int i = 0; i < e; ++i) segBase += counts[i];
  const int cnt = counts[e];

  const int lane = threadIdx.x & 31;
  const int wave = threadIdx.x >> 5;
  const int hl   = lane >> 4;
  const int l16  = lane & 15;
  const int wm   = wave & (WAVES_M - 1);
  const int wn   = wave >> 2;

  const int rowBase = segBase + blockIdx.y * BLOCK_M + wm * (TILES_M * 16);
  const int colBase = blockIdx.x * BLOCK_N + wn * (TILES_N * 16);
  const float* We   = W + (size_t)e * Ntot * K;

  v8f acc[TILES_M][TILES_N];
  const v8f vzero = {};
#pragma unroll
  for (int mt = 0; mt < TILES_M; ++mt)
#pragma unroll
    for (int nt = 0; nt < TILES_N; ++nt) acc[mt][nt] = vzero;

  for (int kk = 0; kk < K; kk += 32) {
    v16bf ahi[TILES_M], alo[TILES_M];
#pragma unroll
    for (int mt = 0; mt < TILES_M; ++mt) {
      int r = rowBase + mt * 16 + l16;
      if (r >= Ttot) r = Ttot - 1;
      load_frag_a(X, K, r, kk, hl, ahi[mt], alo[mt]);
    }
    v16bf bhi[TILES_N], blo[TILES_N];
#pragma unroll
    for (int nt = 0; nt < TILES_N; ++nt) {
      int c = colBase + nt * 16 + l16;
      load_frag_b(We, K, c, kk, hl, bhi[nt], blo[nt]);
    }
#pragma unroll
    for (int mt = 0; mt < TILES_M; ++mt)
#pragma unroll
      for (int nt = 0; nt < TILES_N; ++nt) {
        acc[mt][nt] = __builtin_amdgcn_wmma_f32_16x16x32_bf16(
            false, ahi[mt], false, bhi[nt], (short)0, acc[mt][nt], false, false);
        acc[mt][nt] = __builtin_amdgcn_wmma_f32_16x16x32_bf16(
            false, ahi[mt], false, blo[nt], (short)0, acc[mt][nt], false, false);
        acc[mt][nt] = __builtin_amdgcn_wmma_f32_16x16x32_bf16(
            false, alo[mt], false, bhi[nt], (short)0, acc[mt][nt], false, false);
      }
  }

  const int rowEnd = segBase + cnt;
#pragma unroll
  for (int mt = 0; mt < TILES_M; ++mt)
#pragma unroll
    for (int nt = 0; nt < TILES_N; ++nt) {
      const int col = colBase + nt * 16 + l16;
      const float bv = bias[(size_t)e * Ntot + col];
#pragma unroll
      for (int g = 0; g < 8; ++g) {
        const int row = rowBase + mt * 16 + hl * 8 + g;
        if (row < rowEnd) {
          float v = acc[mt][nt][g] + bv;
          if (act_gelu)
            v = 0.5f * v * (1.0f + erff(v * 0.70710678118654752f));
          Out[(size_t)row * Ntot + col] = v;
        }
      }
    }
}

// ================================ launcher =================================
extern "C" void kernel_launch(void* const* d_in, const int* in_sizes, int n_in,
                              void* d_out, int out_size, void* d_ws, size_t ws_size,
                              hipStream_t stream) {
  (void)in_sizes; (void)n_in; (void)out_size;
  const int E = 8, D = 1024, H = 4096, T = 8192;

  const float* inp = (const float*)d_in[0];   // [T, D]
  const float* w1  = (const float*)d_in[1];   // [E, H, D]
  const float* b1  = (const float*)d_in[2];   // [E, H]
  const float* w2  = (const float*)d_in[3];   // [E, D, H]
  const float* b2  = (const float*)d_in[4];   // [E, D]
  const int*   cnt = (const int*)d_in[5];     // [E]
  float*       out = (float*)d_out;           // [T, D]

  const size_t nX  = (size_t)T * D;           // 8M elems
  const size_t nW1 = (size_t)E * H * D;       // 32M
  const size_t nW2 = (size_t)E * D * H;       // 32M
  const size_t nH  = (size_t)T * H;           // 32M
  // hi/lo bf16 planes: x, w1, w2, h  (2 bytes per plane element)
  const size_t fastBytes = 2 * 2 * (nX + nW1 + nW2 + nH);   // 416 MiB

  dim3 blk(256);
  if (ws_size >= fastBytes) {
    char* p = (char*)d_ws;
    __bf16* xhi  = (__bf16*)p;             p += nX * 2;
    __bf16* xlo  = (__bf16*)p;             p += nX * 2;
    __bf16* w1hi = (__bf16*)p;             p += nW1 * 2;
    __bf16* w1lo = (__bf16*)p;             p += nW1 * 2;
    __bf16* w2hi = (__bf16*)p;             p += nW2 * 2;
    __bf16* w2lo = (__bf16*)p;             p += nW2 * 2;
    __bf16* hhi  = (__bf16*)p;             p += nH * 2;
    __bf16* hlo  = (__bf16*)p;

    split_f32_to_bf16<<<dim3(4096), blk, 0, stream>>>(inp, xhi, xlo, (long long)(nX / 4));
    split_f32_to_bf16<<<dim3(8192), blk, 0, stream>>>(w1, w1hi, w1lo, (long long)(nW1 / 4));
    split_f32_to_bf16<<<dim3(8192), blk, 0, stream>>>(w2, w2hi, w2lo, (long long)(nW2 / 4));

    dim3 g1(H / BLOCK_N, (T / E) / BLOCK_M, E);   // 32 x 8 x 8
    grouped_gemm_bf16x3_pre<<<g1, blk, 0, stream>>>(
        xhi, xlo, w1hi, w1lo, b1, /*OutF=*/nullptr, hhi, hlo, cnt,
        /*K=*/D, /*Ntot=*/H, /*Ttot=*/T, /*gelu=*/1);
    dim3 g2(D / BLOCK_N, (T / E) / BLOCK_M, E);   // 8 x 8 x 8
    grouped_gemm_bf16x3_pre<<<g2, blk, 0, stream>>>(
        hhi, hlo, w2hi, w2lo, b2, /*OutF=*/out, nullptr, nullptr, cnt,
        /*K=*/H, /*Ntot=*/D, /*Ttot=*/T, /*gelu=*/0);
  } else {
    // Fallback: f32 staging for h, in-register hi/lo split (ws >= 128 MiB).
    float* h = (float*)d_ws;
    dim3 g1(H / BLOCK_N, (T / E) / BLOCK_M, E);
    grouped_gemm_bf16x3<<<g1, blk, 0, stream>>>(inp, w1, b1, h, cnt, D, H, T, 1);
    dim3 g2(D / BLOCK_N, (T / E) / BLOCK_M, E);
    grouped_gemm_bf16x3<<<g2, blk, 0, stream>>>(h, w2, b2, out, cnt, H, D, T, 0);
  }
}
